// FusedSVDLSTMCell_70136815943963
// MI455X (gfx1250) — matrix operand order repaired and checked by reading, not compiled, not run on hardware
//
#include <hip/hip_runtime.h>
#include <hip/hip_bf16.h>

typedef __attribute__((ext_vector_type(16))) _Float16 v16h;
typedef __attribute__((ext_vector_type(8)))  _Float16 v8h;
typedef __attribute__((ext_vector_type(8)))  float    v8f;
typedef __attribute__((ext_vector_type(4)))  float    v4f;

#define NB      8192
#define IN_DIM  1024
#define UNITS   1024
#define RANK    64
#define PCOLS   128   // [X@Ku | H@Ru]

static __device__ __forceinline__ v8f wmma_f16(v16h a, v16h b, v8f c) {
  // (neg_a, A, neg_b, B, c_mod, C, reuse_a, reuse_b)
  return __builtin_amdgcn_wmma_f32_16x16x32_f16(false, a, false, b, (short)0, c,
                                                false, false);
}

static __device__ __forceinline__ float fsigmoid(float x) {
  return 1.0f / (1.0f + __expf(-x));
}
static __device__ __forceinline__ float ftanh(float x) {
  float e = __expf(2.0f * x);
  return 1.0f - 2.0f / (e + 1.0f);
}

// -------------------------------------------------------------------------
// Kernel 1: P[:, colbase:colbase+64] = X(8192x1024) @ U(1024x64), f16 out.
// Block = 128 threads (4 waves). Each wave owns one 16-row m-tile (block
// covers 64 rows) and computes all 4 rank n-tiles (4 accumulators).
// U panel (32x64) staged f32->f16 *transposed* in LDS: lds[col][k], so a
// lane's B fragment (rows 16*sel..16*sel+15 of column n) is one contiguous
// 32-byte read -> 2x ds_load_b128.
// -------------------------------------------------------------------------
__global__ __launch_bounds__(128)
void svd_lstm_proj_kernel(const float* __restrict__ X,
                          const float* __restrict__ U,
                          _Float16* __restrict__ P, int colbase) {
  __shared__ alignas(64) _Float16 ldsB[64 * 32];   // [col][krow], 4 KB

  const int tid  = threadIdx.x;
  const int lane = tid & 31;
  const int wave = tid >> 5;
  const int sel  = lane >> 4;          // half-wave select
  const int nl   = lane & 15;
  const int m0   = (blockIdx.x * 4 + wave) * 16;   // wave's m-tile

  const float* xrow = X + (size_t)(m0 + nl) * IN_DIM;

  v8f acc0 = {}, acc1 = {}, acc2 = {}, acc3 = {};

  for (int k0 = 0; k0 < IN_DIM; k0 += 32) {
    // --- cooperative stage of U[k0:k0+32, 0:64] -> transposed f16 panel ---
#pragma unroll
    for (int j = 0; j < 4; ++j) {
      int base = (tid + 128 * j) * 4;              // 0..2044, multiple of 4
      int kr   = base >> 6;
      int kc   = base & 63;
      v4f u4 = *(const v4f*)(U + (size_t)(k0 + kr) * RANK + kc);
      ldsB[(kc + 0) * 32 + kr] = (_Float16)u4[0];
      ldsB[(kc + 1) * 32 + kr] = (_Float16)u4[1];
      ldsB[(kc + 2) * 32 + kr] = (_Float16)u4[2];
      ldsB[(kc + 3) * 32 + kr] = (_Float16)u4[3];
    }
    if (k0 + 32 < IN_DIM) __builtin_prefetch(xrow + k0 + 32, 0, 1);
    __syncthreads();

    // --- A fragment: 16x32 f16, lane holds row m0+nl ---
    // halves 0..7  = K (k0 + 16*sel     .. +7)
    // halves 8..15 = K (k0 + 16*sel + 8 .. +7)
    v16h a;
    {
      const float* plo = xrow + k0 + 16 * sel;
      const float* phi = xrow + k0 + 16 * sel + 8;
      v4f l0 = *(const v4f*)(plo);
      v4f l1 = *(const v4f*)(plo + 4);
      v4f h0 = *(const v4f*)(phi);
      v4f h1 = *(const v4f*)(phi + 4);
#pragma unroll
      for (int c = 0; c < 4; ++c) {
        a[c]      = (_Float16)l0[c];
        a[c + 4]  = (_Float16)l1[c];
        a[c + 8]  = (_Float16)h0[c];
        a[c + 12] = (_Float16)h1[c];
      }
    }

    // --- B fragments: contiguous 32B column reads from transposed panel ---
    v16h b0 = *(const v16h*)(&ldsB[( 0 + nl) * 32 + 16 * sel]);
    v16h b1 = *(const v16h*)(&ldsB[(16 + nl) * 32 + 16 * sel]);
    v16h b2 = *(const v16h*)(&ldsB[(32 + nl) * 32 + 16 * sel]);
    v16h b3 = *(const v16h*)(&ldsB[(48 + nl) * 32 + 16 * sel]);

    acc0 = wmma_f16(a, b0, acc0);
    acc1 = wmma_f16(a, b1, acc1);
    acc2 = wmma_f16(a, b2, acc2);
    acc3 = wmma_f16(a, b3, acc3);
    __syncthreads();
  }

  // store 16x64 f16 strip of P: element (m0 + r + 8*sel, colbase + nt*16 + nl)
#pragma unroll
  for (int r = 0; r < 8; ++r) {
    const size_t row = (size_t)(m0 + r + 8 * sel) * PCOLS + colbase + nl;
    P[row +  0] = (_Float16)acc0[r];
    P[row + 16] = (_Float16)acc1[r];
    P[row + 32] = (_Float16)acc2[r];
    P[row + 48] = (_Float16)acc3[r];
  }
}

// -------------------------------------------------------------------------
// Kernel 2: Z = P(8192x128) @ W(128x4096) + bias, fused LSTM gates.
// W rows 0:64 = kernel_v, rows 64:128 = recurrent_kernel_v.
// Block: 128 threads, owns unit slice u0..u0+15 (covering all 4 gate column
// groups) and a 512-row band. W slice staged once *transposed* in LDS
// (lds[col][k]); B frags for 4 gates x 4 k-steps register-resident.
// -------------------------------------------------------------------------
__global__ __launch_bounds__(128)
void svd_lstm_gate_kernel(const _Float16* __restrict__ P,
                          const float* __restrict__ Kv,
                          const float* __restrict__ Rv,
                          const float* __restrict__ bias,
                          const float* __restrict__ Cprev,
                          float* __restrict__ Hout,
                          float* __restrict__ Cout) {
  __shared__ alignas(64) _Float16 ldsW[64 * PCOLS];   // [col][k], 16 KB

  const int tid  = threadIdx.x;
  const int lane = tid & 31;
  const int wave = tid >> 5;
  const int sel  = lane >> 4;
  const int nl   = lane & 15;
  const int u0      = blockIdx.x * 16;
  const int rowbase = blockIdx.y * 512;

  // --- stage W slice transposed: col group cg = g*16+j -> col g*1024+u0+j
  for (int e = tid; e < PCOLS * 64; e += 128) {
    int k  = e >> 6;
    int cg = e & 63;
    int col = (cg >> 4) * UNITS + u0 + (cg & 15);
    float w = (k < 64) ? Kv[(size_t)k * (4 * UNITS) + col]
                       : Rv[(size_t)(k - 64) * (4 * UNITS) + col];
    ldsW[cg * PCOLS + k] = (_Float16)w;
  }
  __syncthreads();

  // --- build 16 register-resident B fragments: bf[gate][kstep] ---
  // fragment = rows 32*s + 16*sel .. +15 of column g*16+nl -> contiguous 32B
  v16h bf[4][4];
#pragma unroll
  for (int g = 0; g < 4; ++g) {
#pragma unroll
    for (int s = 0; s < 4; ++s) {
      bf[g][s] = *(const v16h*)(&ldsW[(g * 16 + nl) * PCOLS + 32 * s + 16 * sel]);
    }
  }

  const int u = u0 + nl;
  const float bi  = bias[u];
  const float bff = bias[UNITS + u];
  const float bg  = bias[2 * UNITS + u];
  const float bo  = bias[3 * UNITS + u];

  // --- row-tile loop: 32 tiles per block, 8 per wave (uniform, EXEC=all-1s)
  for (int t = wave; t < 32; t += 4) {
    const int m0 = rowbase + t * 16;
    const _Float16* prow = P + (size_t)(m0 + nl) * PCOLS;

    v8f acc0 = {}, acc1 = {}, acc2 = {}, acc3 = {};
#pragma unroll
    for (int s = 0; s < 4; ++s) {
      v16h a;
      v8h lo = *(const v8h*)(prow + 32 * s + 16 * sel);
      v8h hi = *(const v8h*)(prow + 32 * s + 16 * sel + 8);
#pragma unroll
      for (int c = 0; c < 8; ++c) { a[c] = lo[c]; a[c + 8] = hi[c]; }
      acc0 = wmma_f16(a, bf[0][s], acc0);
      acc1 = wmma_f16(a, bf[1][s], acc1);
      acc2 = wmma_f16(a, bf[2][s], acc2);
      acc3 = wmma_f16(a, bf[3][s], acc3);
    }

    // --- fused gates; element (m0 + r + 8*sel, u) per C-layout ---
#pragma unroll
    for (int r = 0; r < 8; ++r) {
      const int m = m0 + 8 * sel + r;
      float zi = acc0[r] + bi;
      float zf = acc1[r] + bff;
      float zg = acc2[r] + bg;
      float zo = acc3[r] + bo;
      float ig = fsigmoid(zi);
      float fg = fsigmoid(zf);
      float gg = ftanh(zg);
      float og = fsigmoid(zo);
      float cp = Cprev[(size_t)m * UNITS + u];
      float cn = fg * cp + ig * gg;
      float hn = og * ftanh(cn);
      Hout[(size_t)m * UNITS + u] = hn;
      Cout[(size_t)m * UNITS + u] = cn;
    }
  }
}

extern "C" void kernel_launch(void* const* d_in, const int* in_sizes, int n_in,
                              void* d_out, int out_size, void* d_ws, size_t ws_size,
                              hipStream_t stream) {
  const float* X    = (const float*)d_in[0];  // inputs            (8192,1024)
  const float* H    = (const float*)d_in[1];  // h_tm1             (8192,1024)
  const float* C    = (const float*)d_in[2];  // c_tm1             (8192,1024)
  const float* Ku   = (const float*)d_in[3];  // kernel_u          (1024,64)
  const float* Kv   = (const float*)d_in[4];  // kernel_v          (64,4096)
  const float* Ru   = (const float*)d_in[5];  // recurrent_kernel_u(1024,64)
  const float* Rv   = (const float*)d_in[6];  // recurrent_kernel_v(64,4096)
  const float* bias = (const float*)d_in[7];  // bias              (4096)

  float* Hout = (float*)d_out;                       // h (8192,1024)
  float* Cout = Hout + (size_t)NB * UNITS;           // c (8192,1024)
  _Float16* P = (_Float16*)d_ws;                     // (8192,128) f16, 2 MB

  dim3 blk(128);
  svd_lstm_proj_kernel<<<dim3(NB / 64), blk, 0, stream>>>(X, Ku, P, 0);
  svd_lstm_proj_kernel<<<dim3(NB / 64), blk, 0, stream>>>(H, Ru, P, 64);
  svd_lstm_gate_kernel<<<dim3(UNITS / 16, NB / 512), blk, 0, stream>>>(
      P, Kv, Rv, bias, C, Hout, Cout);
}
